// FusionExpert_84232898609750
// MI455X (gfx1250) — compile-verified, matrix operand order checked by reading, not run on hardware
//
#include <hip/hip_runtime.h>
#include <hip/hip_bf16.h>
#include <math.h>

typedef __attribute__((ext_vector_type(16))) __bf16 v16bf;
typedef __attribute__((ext_vector_type(8)))  float  v8f;

// Problem constants (from reference setup_inputs)
static constexpr int kE  = 8;
static constexpr int kT  = 2048;
static constexpr int kH  = 768;
static constexpr int kI  = 3072;
static constexpr int kTM = 32;    // tokens per workgroup
static constexpr int kIC = 256;   // I-chunk width per pass

// LDS layout (bytes) for main kernel:
//  [0     , 49152)  Xbf : [kTM][kH]  bf16  (48 KB) } reused at epilogue as
//  [49152 , 65536)  Ibf : [kTM][kIC] bf16  (16 KB) } Ybuf f32 [kTM][kH] (96 KB)
static constexpr int LDS_X     = 0;
static constexpr int LDS_INTER = 49152;
static constexpr int LDS_TOTAL = 98304;

// ---------------------------------------------------------------------------
// Prep: fp32 [E][R][C] -> bf16 [E][C][R] (transpose + convert), LDS-tiled so
// both global read and global write are coalesced. Runs once per launch; the
// 75.5 MB bf16 weight image stays resident in the 192 MB L2.
// ---------------------------------------------------------------------------
__global__ __launch_bounds__(256)
void transpose_to_bf16_kernel(const float* __restrict__ src,
                              __bf16* __restrict__ dst, int R, int C)
{
    __shared__ __bf16 tile[32][33];
    const int e  = blockIdx.z;
    const int c0 = blockIdx.x * 32;
    const int r0 = blockIdx.y * 32;
    const float* s = src + (long long)e * R * C;
    __bf16*      d = dst + (long long)e * C * R;
    const int lx = threadIdx.x & 31;
    const int ly = threadIdx.x >> 5;
#pragma unroll
    for (int i = 0; i < 4; ++i) {
        int r = ly + i * 8;
        tile[r][lx] = (__bf16)s[(long long)(r0 + r) * C + c0 + lx];
    }
    __syncthreads();
#pragma unroll
    for (int i = 0; i < 4; ++i) {
        int c = ly + i * 8;
        d[(long long)(c0 + c) * R + r0 + lx] = tile[lx][c];
    }
}

// ---------------------------------------------------------------------------
// Fused MoE FFN: x@W1 -> +b1 -> exact GELU -> @W2 -> +b2 -> +residual -> LN.
// One workgroup = (expert, 32-token tile); 8 wave32 waves.
//  - A fragments (X, intermediate) from LDS: shared by all 8 waves.
//  - B fragments: uniform (SGPR) base per chunk + per-lane 32-bit byte
//    offsets; fully-unrolled k-loops put the k-advance in the load's 24-bit
//    immediate -> global_load_b128 saddr+voffset form, no pointer PHIs,
//    no FLAT path. Weights stay L2/WGP$-resident.
// ---------------------------------------------------------------------------
__global__ __launch_bounds__(256)
void moe_ffn_wmma_kernel(const float* __restrict__ x,
                         const __bf16* __restrict__ w1t,  // [E][I][H] bf16
                         const float* __restrict__ b1,
                         const __bf16* __restrict__ w2t,  // [E][H][I] bf16
                         const float* __restrict__ b2,
                         const float* __restrict__ gamma,
                         const float* __restrict__ beta,
                         float* __restrict__ out)
{
    extern __shared__ char smem[];
    __bf16* Xbf  = (__bf16*)(smem + LDS_X);
    __bf16* Ibf  = (__bf16*)(smem + LDS_INTER);
    float*  Ybuf = (float*)(smem);            // spans both regions (96 KB)

    const int tid  = threadIdx.x;
    const int lane = tid & 31;
    const int wv   = tid >> 5;                 // 0..7
    const int tilesPerExp = kT / kTM;          // 64
    const int e    = blockIdx.x / tilesPerExp;
    const int m0   = (blockIdx.x % tilesPerExp) * kTM;
    const long long rowbase = (long long)e * kT + m0;

    const float*  Xg   = x   + rowbase * kH;
    const __bf16* W1Te = w1t + (long long)e * kI * kH;  // row n in [0,I), stride H
    const __bf16* W2Te = w2t + (long long)e * kH * kI;  // row n in [0,H), stride I
    const float*  B1g  = b1 + e * kI;
    const float*  B2g  = b2 + e * kH;

    // ---- Stage X tile to LDS as bf16 (row-major [kTM][kH]) ----
    for (int i = tid; i < kTM * kH; i += 256)
        Xbf[i] = (__bf16)Xg[i];
    __syncthreads();

    // WMMA fragment addressing (wave32; this code's contiguous k order, same for A and B):
    //  A (16x32 bf16): lane holds row M=lane%16, contiguous K-half by lane/16
    //  B (32x16 bf16): lane holds col N=lane%16, contiguous K-half by lane/16
    const int arow  = lane & 15;
    const int khalf = (lane >> 4) * 16;

    // LDS A-fragment bases (immediate ds offsets cover the k dimension)
    const __bf16* aX0 = Xbf + arow * kH + khalf;            // mt=0
    const __bf16* aX1 = Xbf + (16 + arow) * kH + khalf;     // mt=1
    const __bf16* aI0 = Ibf + arow * kIC + khalf;
    const __bf16* aI1 = Ibf + (16 + arow) * kIC + khalf;

    // Persistent GEMM2 accumulators: this wave owns out cols [wv*96, wv*96+96)
    const int n2base = wv * 96;
    v8f acc2[2][6] = {};

    // Per-lane BYTE offsets of each B row (computed once; 32-bit VGPRs)
    unsigned rb2b[6];
#pragma unroll
    for (int nt = 0; nt < 6; ++nt)
        rb2b[nt] = ((unsigned)(n2base + nt * 16 + arow) * (unsigned)kI + khalf) * 2u;
    unsigned rb1b[2];
#pragma unroll
    for (int nt = 0; nt < 2; ++nt)
        rb1b[nt] = ((unsigned)(wv * 32 + nt * 16 + arow) * (unsigned)kH + khalf) * 2u;

    for (int ic = 0; ic < kI / kIC; ++ic) {
        // Uniform per-chunk bases (SALU only; derived directly from kernel args)
        const char* base1 = (const char*)W1Te + (size_t)ic * kIC * kH * 2;  // chunk rows
        const char* base2 = (const char*)W2Te + (size_t)ic * kIC * 2;       // chunk K cols

        // ============ GEMM1: inter[32x256] = X @ W1[:, chunk] ===============
        v8f acc1[2][2] = {};
#pragma unroll
        for (int k0 = 0; k0 < kH; k0 += 32) {       // fully unrolled, imm offsets
            v16bf a0  = *(const v16bf*)(aX0 + k0);
            v16bf a1  = *(const v16bf*)(aX1 + k0);
            v16bf b0  = *(const v16bf*)(base1 + rb1b[0] + k0 * 2);
            v16bf b1v = *(const v16bf*)(base1 + rb1b[1] + k0 * 2);
            if ((k0 & 32) == 0 && k0 + 64 < kH) {   // one 128B line ahead
                __builtin_prefetch(base1 + rb1b[0] + k0 * 2 + 128, 0, 3);
                __builtin_prefetch(base1 + rb1b[1] + k0 * 2 + 128, 0, 3);
            }
            acc1[0][0] = __builtin_amdgcn_wmma_f32_16x16x32_bf16(
                false, a0, false, b0,  (short)0, acc1[0][0], false, false);
            acc1[1][0] = __builtin_amdgcn_wmma_f32_16x16x32_bf16(
                false, a1, false, b0,  (short)0, acc1[1][0], false, false);
            acc1[0][1] = __builtin_amdgcn_wmma_f32_16x16x32_bf16(
                false, a0, false, b1v, (short)0, acc1[0][1], false, false);
            acc1[1][1] = __builtin_amdgcn_wmma_f32_16x16x32_bf16(
                false, a1, false, b1v, (short)0, acc1[1][1], false, false);
        }

        // bias + exact GELU -> bf16 intermediate tile in LDS.
        // C layout: lane holds col N=lane%16; VGPR r holds row M=(lane/16)*8+r
#pragma unroll
        for (int mt = 0; mt < 2; ++mt) {
            int mrow = mt * 16 + (lane >> 4) * 8;
#pragma unroll
            for (int nt = 0; nt < 2; ++nt) {
                int c = wv * 32 + nt * 16 + (lane & 15);
                float bias = B1g[ic * kIC + c];
#pragma unroll
                for (int r = 0; r < 8; ++r) {
                    float v  = acc1[mt][nt][r] + bias;
                    float gl = 0.5f * v * (1.0f + erff(v * 0.70710678118654752f));
                    Ibf[(mrow + r) * kIC + c] = (__bf16)gl;
                }
            }
        }
        __syncthreads();   // Ibf ready for all waves

        // ============ GEMM2 partial: out += inter[32x256] @ W2[chunk, :] ====
#pragma unroll
        for (int k0 = 0; k0 < kIC; k0 += 32) {      // fully unrolled, imm offsets
            v16bf a0 = *(const v16bf*)(aI0 + k0);
            v16bf a1 = *(const v16bf*)(aI1 + k0);
            v16bf b[6];
#pragma unroll
            for (int nt = 0; nt < 6; ++nt) {
                b[nt] = *(const v16bf*)(base2 + rb2b[nt] + k0 * 2);
                if ((k0 & 32) == 0)
                    __builtin_prefetch(base2 + rb2b[nt] + k0 * 2 + 128, 0, 3);
            }
#pragma unroll
            for (int nt = 0; nt < 6; ++nt) {
                acc2[0][nt] = __builtin_amdgcn_wmma_f32_16x16x32_bf16(
                    false, a0, false, b[nt], (short)0, acc2[0][nt], false, false);
                acc2[1][nt] = __builtin_amdgcn_wmma_f32_16x16x32_bf16(
                    false, a1, false, b[nt], (short)0, acc2[1][nt], false, false);
            }
        }
        __syncthreads();   // all waves done with Ibf before next chunk rewrites
    }

    // ======= Epilogue: y = out + b2 + residual -> Ybuf (f32, 96 KB) =========
#pragma unroll
    for (int mt = 0; mt < 2; ++mt) {
        int mrow = mt * 16 + (lane >> 4) * 8;
#pragma unroll
        for (int nt = 0; nt < 6; ++nt) {
            int c = n2base + nt * 16 + (lane & 15);
            float bias = B2g[c];
#pragma unroll
            for (int r = 0; r < 8; ++r) {
                int m = mrow + r;
                Ybuf[m * kH + c] = acc2[mt][nt][r] + bias + Xg[(long long)m * kH + c];
            }
        }
    }
    __syncthreads();

    // ======= LayerNorm over H=768: wave wv handles 4 rows ===================
    for (int rr = 0; rr < 4; ++rr) {
        int m = wv * 4 + rr;
        float s = 0.f, ss = 0.f;
        float vals[kH / 32];
#pragma unroll
        for (int j = 0; j < kH / 32; ++j) {
            float v = Ybuf[m * kH + lane + j * 32];
            vals[j] = v;
            s  += v;
            ss += v * v;
        }
#pragma unroll
        for (int off = 16; off > 0; off >>= 1) {
            s  += __shfl_xor(s,  off, 32);
            ss += __shfl_xor(ss, off, 32);
        }
        float mean = s * (1.0f / kH);
        float var  = ss * (1.0f / kH) - mean * mean;
        float rstd = rsqrtf(var + 1e-12f);
        float* Og = out + (rowbase + m) * kH;
#pragma unroll
        for (int j = 0; j < kH / 32; ++j) {
            int c = lane + j * 32;
            Og[c] = (vals[j] - mean) * rstd * gamma[c] + beta[c];
        }
    }
}

extern "C" void kernel_launch(void* const* d_in, const int* in_sizes, int n_in,
                              void* d_out, int out_size, void* d_ws, size_t ws_size,
                              hipStream_t stream) {
    (void)in_sizes; (void)n_in; (void)ws_size; (void)out_size;
    const float* x     = (const float*)d_in[0];
    const float* w1    = (const float*)d_in[1];
    const float* b1    = (const float*)d_in[2];
    const float* w2    = (const float*)d_in[3];
    const float* b2    = (const float*)d_in[4];
    const float* gamma = (const float*)d_in[5];
    const float* beta  = (const float*)d_in[6];
    float* out = (float*)d_out;

    // Workspace: bf16 pre-transposed weights (75.5 MB total)
    __bf16* w1t = (__bf16*)d_ws;                              // [E][I][H]
    __bf16* w2t = w1t + (size_t)kE * kH * kI;                 // [E][H][I]

    // Prep pass: w1 [E][H][I] -> w1t [E][I][H] ; w2 [E][I][H] -> w2t [E][H][I]
    {
        dim3 blk(256);
        dim3 g1(kI / 32, kH / 32, kE);
        transpose_to_bf16_kernel<<<g1, blk, 0, stream>>>(w1, w1t, kH, kI);
        dim3 g2(kH / 32, kI / 32, kE);
        transpose_to_bf16_kernel<<<g2, blk, 0, stream>>>(w2, w2t, kI, kH);
    }

    // Main fused pass
    dim3 grid(kE * (kT / kTM));   // 512 workgroups
    dim3 block(256);              // 8 wave32 waves
    moe_ffn_wmma_kernel<<<grid, block, LDS_TOTAL, stream>>>(
        x, w1t, b1, w2t, b2, gamma, beta, out);
}
